// MyHypergraphConv_15556371546537
// MI455X (gfx1250) — compile-verified, matrix operand order checked by reading, not run on hardware
//
#include <hip/hip_runtime.h>
#include <stdint.h>

// ---------------------------------------------------------------------------
// Problem constants (fixed by the reference)
// ---------------------------------------------------------------------------
#define N_NODES 16384
#define N_EDGES 8192
#define DIN 128
#define DOUT 128

typedef __attribute__((ext_vector_type(16))) __bf16 v16bf;
typedef __attribute__((ext_vector_type(2)))  __bf16 v2bf;
typedef __attribute__((ext_vector_type(2)))  float  v2f;
typedef __attribute__((ext_vector_type(8)))  float  v8f;

// Packed f32->bf16 (RNE) — lets the compiler use v_cvt_pk_bf16_f32.
__device__ __forceinline__ unsigned int packbf2(float a, float b) {
  v2f f; f.x = a; f.y = b;
  v2bf h = __builtin_convertvector(f, v2bf);
  return __builtin_bit_cast(unsigned int, h);
}
// Scalar fallback for strided single-element stores (RNE, same rounding).
__device__ __forceinline__ unsigned short bf16bits(float f) {
  unsigned int u = __float_as_uint(f);
  unsigned int r = u + 0x7FFFu + ((u >> 16) & 1u);
  return (unsigned short)(r >> 16);
}

// ---------------------------------------------------------------------------
// Kernel 1: one pass over H -> per-block node-degree partials + full dei.
// Block owns 32 full columns (grid = E/32 = 256). Wave w sweeps rows w, w+8,
// ...; per-row wave reduction -> row partial written once (no atomics,
// deterministic). Column sums finished in-block -> dei written directly.
// ---------------------------------------------------------------------------
__global__ __launch_bounds__(256) void deg_kernel(const float* __restrict__ H,
                                                  float* __restrict__ nodepart,
                                                  float* __restrict__ dei) {
  const int wave = threadIdx.x >> 5;
  const int lane = threadIdx.x & 31;
  const int c = blockIdx.x * 32 + lane;
  float cacc = 0.f;
  __shared__ float cs[8][32];
  for (int r = wave; r < N_NODES; r += 8) {
    float v = H[(size_t)r * N_EDGES + c];
    cacc += v;
    float s = v;
#pragma unroll
    for (int off = 16; off > 0; off >>= 1) s += __shfl_xor(s, off, 32);
    if (lane == 0) nodepart[(size_t)blockIdx.x * N_NODES + r] = s;
  }
  cs[wave][lane] = cacc;
  __syncthreads();
  if (wave == 0) {
    float t = 0.f;
#pragma unroll
    for (int w = 0; w < 8; ++w) t += cs[w][lane];      // fixed order
    dei[c] = (t > 0.f) ? (1.f / t) : 0.f;
  }
}

// ---------------------------------------------------------------------------
// Kernel 2: deterministic reduce of the 256 row partials -> dvih = deg^-1/2
// ---------------------------------------------------------------------------
__global__ __launch_bounds__(256) void dvih_kernel(const float* __restrict__ nodepart,
                                                   float* __restrict__ dvih) {
  const int n = blockIdx.x * 256 + threadIdx.x;
  float s = 0.f;
  for (int b = 0; b < N_EDGES / 32; ++b)
    s += nodepart[(size_t)b * N_NODES + n];
  dvih[n] = (s > 0.f) ? rsqrtf(s) : 0.f;
}

// ---------------------------------------------------------------------------
// Kernel 3: xsT[d][n] = bf16(dvih[n] * x[n][d])  (LDS 32x32 tile transpose)
// ---------------------------------------------------------------------------
__global__ __launch_bounds__(256) void xsT_kernel(const float* __restrict__ x,
                                                  const float* __restrict__ dvih,
                                                  unsigned short* __restrict__ xsT) {
  __shared__ float t[32][33];
  const int tx = threadIdx.x & 31;
  const int ty = threadIdx.x >> 5;
  const int n0 = blockIdx.x * 32;
  const int d0 = blockIdx.y * 32;
#pragma unroll
  for (int j = 0; j < 4; ++j) {
    const int n = n0 + ty + 8 * j;
    t[ty + 8 * j][tx] = x[(size_t)n * DIN + d0 + tx] * dvih[n];
  }
  __syncthreads();
#pragma unroll
  for (int j = 0; j < 4; ++j) {
    const int d = d0 + ty + 8 * j;
    xsT[(size_t)d * N_NODES + n0 + tx] = bf16bits(t[tx][ty + 8 * j]);
  }
}

// ---------------------------------------------------------------------------
// Unified bf16 WMMA GEMM:  C[M,N] = A[M,K] * B[K,N]  (+ epilogue)
//   MODE 0 (GEMM1): A = bf16 xsT, B = f32 H in [k][n] order,
//                   C = f32, epilogue: * scale[n] (dei)
//   MODE 1 (GEMM2): A = f32 H,   B = f32 YedgeT in [n][k] order,
//                   C = bf16, epilogue: * scale[m] (dvih)
//   MODE 2 (GEMM3): A = bf16 G,  B = f32 W in [n][k] order,
//                   C = f32, epilogue: + scale[n] (bias)
// Block tile 128x64, BK=32, 8 waves (4x2), wave = 32x32 sub-tile via 2x2
// v_wmma_f32_16x16x32_bf16 accumulators. Double-buffered LDS: global loads
// for tile i+1 issue before the WMMAs of tile i; one barrier per K-step.
// ---------------------------------------------------------------------------
constexpr int BM = 128, BN = 64, BK = 32;
constexpr int SA = BK + 8;   // LDS row stride in halfs (80B) -> bank skew
constexpr int SB = BK + 8;

template <int MODE>
__global__ __launch_bounds__(256) void gemm_bf16(
    const float* __restrict__ Af32, const unsigned short* __restrict__ Abf,
    const float* __restrict__ Bsrc, const float* __restrict__ scale,
    float* __restrict__ Cf32, unsigned short* __restrict__ Cbf,
    int K, int lda, int ldb, int ldc) {
  __shared__ unsigned short As[2][BM * SA];   // [m][k] bf16
  __shared__ unsigned short Bs[2][BN * SB];   // [n][k] bf16

  const int tid  = threadIdx.x;
  const int wave = tid >> 5;
  const int lane = tid & 31;
  const int wy = wave >> 1;     // 0..3 -> M offset
  const int wx = wave & 1;      // 0..1 -> N offset
  const int h  = lane >> 4;     // ISA half-wave select
  const int ln = lane & 15;

  const int m0 = blockIdx.y * BM;
  const int n0 = blockIdx.x * BN;

  // staging coordinates (fixed per thread)
  const int arow = tid >> 1;            // 0..127
  const int akb  = (tid & 1) * 16;      // 0 / 16
  const int bk   = tid >> 3;            // MODE0: 0..31
  const int bnb  = (tid & 7) * 8;       // MODE0: 0..56
  const int bn   = tid >> 2;            // NMAJOR: 0..63
  const int bkb  = (tid & 3) * 8;       // NMAJOR: 0..24

  unsigned int aR[8];                   // 16 halfs staged in regs
  unsigned int bR[4];                   // 8 halfs staged in regs

  auto loadTile = [&](int k0) {
    if (MODE == 0 || MODE == 2) {       // A already bf16
      const unsigned short* src = Abf + (size_t)(m0 + arow) * lda + (k0 + akb);
      uint4 q0 = *(const uint4*)(src);
      uint4 q1 = *(const uint4*)(src + 8);
      aR[0] = q0.x; aR[1] = q0.y; aR[2] = q0.z; aR[3] = q0.w;
      aR[4] = q1.x; aR[5] = q1.y; aR[6] = q1.z; aR[7] = q1.w;
    } else {                            // A f32 -> packed bf16 convert
      const float* src = Af32 + (size_t)(m0 + arow) * lda + (k0 + akb);
#pragma unroll
      for (int j = 0; j < 4; ++j) {
        float4 f = *(const float4*)(src + 4 * j);
        aR[2 * j]     = packbf2(f.x, f.y);
        aR[2 * j + 1] = packbf2(f.z, f.w);
      }
    }
    if (MODE == 0) {                    // B source [k][n] (rows of H)
      const float* src = Bsrc + (size_t)(k0 + bk) * ldb + (n0 + bnb);
      float4 f0 = ((const float4*)src)[0];
      float4 f1 = ((const float4*)src)[1];
      bR[0] = packbf2(f0.x, f0.y); bR[1] = packbf2(f0.z, f0.w);
      bR[2] = packbf2(f1.x, f1.y); bR[3] = packbf2(f1.z, f1.w);
    } else {                            // B source [n][k]
      const float* src = Bsrc + (size_t)(n0 + bn) * ldb + (k0 + bkb);
      float4 f0 = ((const float4*)src)[0];
      float4 f1 = ((const float4*)src)[1];
      bR[0] = packbf2(f0.x, f0.y); bR[1] = packbf2(f0.z, f0.w);
      bR[2] = packbf2(f1.x, f1.y); bR[3] = packbf2(f1.z, f1.w);
    }
  };

  auto storeTile = [&](int buf) {
    *(uint4*)&As[buf][arow * SA + akb]     = make_uint4(aR[0], aR[1], aR[2], aR[3]);
    *(uint4*)&As[buf][arow * SA + akb + 8] = make_uint4(aR[4], aR[5], aR[6], aR[7]);
    if (MODE == 0) {                    // transpose into Bs[n][k]
#pragma unroll
      for (int j = 0; j < 4; ++j) {
        Bs[buf][(bnb + 2 * j)     * SB + bk] = (unsigned short)(bR[j]);
        Bs[buf][(bnb + 2 * j + 1) * SB + bk] = (unsigned short)(bR[j] >> 16);
      }
    } else {
      *(uint2*)&Bs[buf][bn * SB + bkb]     = make_uint2(bR[0], bR[1]);
      *(uint2*)&Bs[buf][bn * SB + bkb + 4] = make_uint2(bR[2], bR[3]);
    }
  };

  v8f acc[2][2] = {};
  const int nk = K / BK;

  loadTile(0);
  storeTile(0);
  __syncthreads();

  for (int kt = 0; kt < nk; ++kt) {
    const int cur = kt & 1;
    const bool more = (kt + 1) < nk;
    if (more) loadTile((kt + 1) * BK);  // global loads overlap the WMMAs

    // Fragment loads: per ISA 16-bit layout, lane (h,ln) takes K chunks
    // [8h,8h+8) and [16+8h,16+8h+8) of its row -> two ds_load_b128 each.
    union Frag { uint4 q[2]; v16bf v; };
    Frag fa[2], fb[2];
#pragma unroll
    for (int r = 0; r < 2; ++r) {
      const int ar = wy * 32 + r * 16 + ln;
      fa[r].q[0] = *(const uint4*)&As[cur][ar * SA + 8 * h];
      fa[r].q[1] = *(const uint4*)&As[cur][ar * SA + 16 + 8 * h];
    }
#pragma unroll
    for (int c = 0; c < 2; ++c) {
      const int br = wx * 32 + c * 16 + ln;
      fb[c].q[0] = *(const uint4*)&Bs[cur][br * SB + 8 * h];
      fb[c].q[1] = *(const uint4*)&Bs[cur][br * SB + 16 + 8 * h];
    }
#pragma unroll
    for (int r = 0; r < 2; ++r)
#pragma unroll
      for (int c = 0; c < 2; ++c)
        acc[r][c] = __builtin_amdgcn_wmma_f32_16x16x32_bf16(
            false, fa[r].v, false, fb[c].v, (short)0, acc[r][c], false, false);

    if (more) storeTile(cur ^ 1);       // write-only the other buffer
    __syncthreads();                    // single barrier per K-step
  }

  // ---- epilogue: C vgpr v holds (M = 8h + v, N = ln) of its 16x16 tile ----
#pragma unroll
  for (int r = 0; r < 2; ++r) {
#pragma unroll
    for (int c = 0; c < 2; ++c) {
      const int gn = n0 + wx * 32 + c * 16 + ln;
      const int mb = m0 + wy * 32 + r * 16 + 8 * h;
#pragma unroll
      for (int v = 0; v < 8; ++v) {
        const int gm = mb + v;
        const float val = acc[r][c][v];
        if (MODE == 0) {
          Cf32[(size_t)gm * ldc + gn] = val * scale[gn];          // * dei[e]
        } else if (MODE == 1) {
          Cbf[(size_t)gm * ldc + gn] = bf16bits(val * scale[gm]); // * dvih[n]
        } else {
          Cf32[(size_t)gm * ldc + gn] = val + scale[gn];          // + b
        }
      }
    }
  }
}

// ---------------------------------------------------------------------------
// Host-side launch
// ---------------------------------------------------------------------------
extern "C" void kernel_launch(void* const* d_in, const int* in_sizes, int n_in,
                              void* d_out, int out_size, void* d_ws, size_t ws_size,
                              hipStream_t stream) {
  (void)in_sizes; (void)n_in; (void)out_size; (void)ws_size;
  const float* x = (const float*)d_in[0];
  const float* H = (const float*)d_in[1];
  const float* W = (const float*)d_in[2];
  const float* b = (const float*)d_in[3];
  float* out = (float*)d_out;

  char* ws = (char*)d_ws;
  size_t off = 0;
  float* nodepart = (float*)(ws + off); off += (size_t)(N_EDGES / 32) * N_NODES * 4; // 16 MB
  float* dvih = (float*)(ws + off);     off += (size_t)N_NODES * 4;
  float* dei  = (float*)(ws + off);     off += (size_t)N_EDGES * 4;
  unsigned short* xsT = (unsigned short*)(ws + off); off += (size_t)DIN * N_NODES * 2; // 4 MB
  float* YedgeT = (float*)(ws + off);   off += (size_t)DIN * N_EDGES * 4;              // 4 MB
  unsigned short* G = (unsigned short*)(ws + off); off += (size_t)N_NODES * DOUT * 2;  // 4 MB

  // 1) degrees (one pass over H), deterministic
  deg_kernel<<<N_EDGES / 32, 256, 0, stream>>>(H, nodepart, dei);
  dvih_kernel<<<N_NODES / 256, 256, 0, stream>>>(nodepart, dvih);
  // 2) xsT = (dvih * x)^T  in bf16
  xsT_kernel<<<dim3(N_NODES / 32, DIN / 32), 256, 0, stream>>>(x, dvih, xsT);
  // 3) YedgeT[128,E] = xsT[128,N] * H[N,E], columns scaled by dei
  gemm_bf16<0><<<dim3(N_EDGES / BN, DIN / BM), 256, 0, stream>>>(
      nullptr, xsT, H, dei, YedgeT, nullptr,
      /*K=*/N_NODES, /*lda=*/N_NODES, /*ldb=*/N_EDGES, /*ldc=*/N_EDGES);
  // 4) G[N,128] = H[N,E] * (dei∘Yedge), rows scaled by dvih, bf16 output
  gemm_bf16<1><<<dim3(DIN / BN, N_NODES / BM), 256, 0, stream>>>(
      H, nullptr, YedgeT, dvih, nullptr, G,
      /*K=*/N_EDGES, /*lda=*/N_EDGES, /*ldb=*/N_EDGES, /*ldc=*/DIN);
  // 5) out[N,128] = G * W^T + b
  gemm_bf16<2><<<dim3(DOUT / BN, N_NODES / BM), 256, 0, stream>>>(
      nullptr, G, W, b, out, nullptr,
      /*K=*/DIN, /*lda=*/DIN, /*ldb=*/DIN, /*ldc=*/DOUT);
}
